// GCN_79766132621935
// MI455X (gfx1250) — matrix operand order repaired and checked by reading, not compiled, not run on hardware
//
#include <hip/hip_runtime.h>
#include <hip/hip_bf16.h>

#define N_NODES   40000
#define N_EDGES   640000
#define HID       128
#define N_GRAPHS  256
#define N_CLASSES 5
#define TOT_E     (N_EDGES + N_NODES)
#define NH        (N_NODES * HID)

typedef __attribute__((ext_vector_type(16))) __bf16 v16bf;
typedef __attribute__((ext_vector_type(8)))  __bf16 v8bf;
typedef __attribute__((ext_vector_type(8)))  float  v8f;

__device__ __forceinline__ __bf16 f2bf(float f) {
    unsigned u = __builtin_bit_cast(unsigned, f);
    unsigned r = (u + 0x7FFFu + ((u >> 16) & 1u)) >> 16;   // round-to-nearest-even
    unsigned short s = (unsigned short)r;
    return __builtin_bit_cast(__bf16, s);
}

// ---------------- weight f32 -> bf16 ----------------
__global__ __launch_bounds__(256) void cvt_w_kernel(const float* __restrict__ W,
                                                    __bf16* __restrict__ Wb, int n) {
    int i = blockIdx.x * 256 + threadIdx.x;
    if (i < n) Wb[i] = f2bf(W[i]);
}

// ---------------- WMMA GEMM: C[40000x128] = A[40000x128] * Wb[128x128] ----------------
// block = 256 threads = 8 waves; block owns 16 rows; wave w owns cols [16w,16w+16)
__global__ __launch_bounds__(256) void gemm_wmma_kernel(const float* __restrict__ A,
                                                        const __bf16* __restrict__ Wb,
                                                        float* __restrict__ C) {
    __shared__ __bf16 Asl[16][HID + 8];          // padded: row stride 136 (16B-aligned chunks)
    const int row0 = blockIdx.x * 16;
    const int tid  = threadIdx.x;

    // stage + convert 16x128 A tile into LDS
    for (int i = tid; i < 16 * HID; i += 256) {
        int r = i >> 7, c = i & (HID - 1);
        Asl[r][c] = f2bf(A[(size_t)(row0 + r) * HID + c]);
    }
    __syncthreads();

    const int wave = tid >> 5;
    const int lane = tid & 31;
    const int r = lane & 15;       // A row / B column / D column
    const int g = lane >> 4;       // lane group
    const int n0 = wave * 16;

    __builtin_prefetch(Wb + n0, 0, 3);           // global_prefetch_b8

    v8f acc = {};
#pragma unroll
    for (int k0 = 0; k0 < HID; k0 += 32) {
        // A 16x32 bf16 layout: lane r row, VGPR0-3 -> K=(0..7)+8g, VGPR4-7 -> K=(16..23)+8g
        v8bf alo = *(const v8bf*)&Asl[r][k0 + 8 * g];
        v8bf ahi = *(const v8bf*)&Asl[r][k0 + 8 * g + 16];
        v16bf a = __builtin_shufflevector(alo, ahi, 0,1,2,3,4,5,6,7,8,9,10,11,12,13,14,15);
        // B 32x16 bf16: lane r column n0+r, elem s -> K=s+8g (s<8), K=s+8+8g (s>=8)
        v16bf b;
#pragma unroll
        for (int s = 0; s < 16; ++s)
            b[s] = Wb[(size_t)(k0 + 8 * g + (s < 8 ? s : s + 8)) * HID + n0 + r];
        acc = __builtin_amdgcn_wmma_f32_16x16x32_bf16(false, a, false, b,
                                                      (short)0, acc, false, false);
    }

    // D 16x16 f32: elem i -> M = i + 8g, N = r
#pragma unroll
    for (int i = 0; i < 8; ++i)
        C[(size_t)(row0 + i + 8 * g) * HID + n0 + r] = acc[i];
}

// ---------------- degree (with self loops) ----------------
__global__ __launch_bounds__(256) void deg_kernel(const int* __restrict__ dst,
                                                  float* __restrict__ deg) {
    int e = blockIdx.x * 256 + threadIdx.x;
    if (e >= TOT_E) return;
    int d = (e < N_EDGES) ? dst[e] : (e - N_EDGES);
    atomicAdd(&deg[d], 1.0f);
}

__global__ __launch_bounds__(256) void dinv_kernel(const float* __restrict__ deg,
                                                   float* __restrict__ dinv) {
    int n = blockIdx.x * 256 + threadIdx.x;
    if (n >= N_NODES) return;
    float d = deg[n];
    dinv[n] = (d > 0.0f) ? rsqrtf(d) : 0.0f;
}

// ---------------- edge scatter: agg[dst] += dinv[s]*dinv[d] * xw[src] ----------------
// one wave32 per edge, 4 features per lane
__global__ __launch_bounds__(256) void aggregate_kernel(const int* __restrict__ src,
                                                        const int* __restrict__ dst,
                                                        const float* __restrict__ dinv,
                                                        const float* __restrict__ xw,
                                                        float* __restrict__ agg) {
    int e = blockIdx.x * 8 + (threadIdx.x >> 5);
    if (e >= TOT_E) return;
    int lane = threadIdx.x & 31;
    int s, d;
    if (e < N_EDGES) { s = src[e]; d = dst[e]; } else { s = d = e - N_EDGES; }
    float w = dinv[s] * dinv[d];
    float4 v = ((const float4*)(xw + (size_t)s * HID))[lane];
    float* p = agg + (size_t)d * HID + lane * 4;
    atomicAdd(p + 0, w * v.x);
    atomicAdd(p + 1, w * v.y);
    atomicAdd(p + 2, w * v.z);
    atomicAdd(p + 3, w * v.w);
}

// ---------------- bias (+ optional relu), in place ----------------
__global__ __launch_bounds__(256) void bias_act_kernel(float* __restrict__ h,
                                                       const float* __restrict__ b,
                                                       int do_relu) {
    int i = blockIdx.x * 256 + threadIdx.x;
    if (i >= NH) return;
    float v = h[i] + b[i & (HID - 1)];
    h[i] = do_relu ? fmaxf(v, 0.0f) : v;
}

// ---------------- mean-pool (sum + count via atomics) ----------------
__global__ __launch_bounds__(256) void pool_kernel(const float* __restrict__ h,
                                                   const int* __restrict__ batch,
                                                   float* __restrict__ pooled,
                                                   float* __restrict__ cnt) {
    int n = blockIdx.x * 8 + (threadIdx.x >> 5);
    if (n >= N_NODES) return;
    int lane = threadIdx.x & 31;
    int b = batch[n];
    float4 v = ((const float4*)(h + (size_t)n * HID))[lane];
    float* p = pooled + (size_t)b * HID + lane * 4;
    atomicAdd(p + 0, v.x);
    atomicAdd(p + 1, v.y);
    atomicAdd(p + 2, v.z);
    atomicAdd(p + 3, v.w);
    if (lane == 0) atomicAdd(&cnt[b], 1.0f);
}

// ---------------- (sums/cnt) @ Wl + bl ----------------
__global__ __launch_bounds__(256) void head_kernel(const float* __restrict__ pooled,
                                                   const float* __restrict__ cnt,
                                                   const float* __restrict__ Wl,
                                                   const float* __restrict__ bl,
                                                   float* __restrict__ out) {
    int idx = blockIdx.x * 256 + threadIdx.x;
    if (idx >= N_GRAPHS * N_CLASSES) return;
    int gph = idx / N_CLASSES, c = idx % N_CLASSES;
    float inv = 1.0f / fmaxf(cnt[gph], 1.0f);
    float acc = 0.0f;
    const float* pr = pooled + (size_t)gph * HID;
#pragma unroll 4
    for (int k = 0; k < HID; ++k) acc += pr[k] * Wl[k * N_CLASSES + c];
    out[idx] = acc * inv + bl[c];
}

// ---------------- host orchestration ----------------
static void run_layer(const float* hin, const float* W, const float* b,
                      float* tmp_xw, float* hout, __bf16* Wb,
                      const int* src, const int* dst, const float* dinv,
                      int do_relu, hipStream_t stream) {
    cvt_w_kernel<<<(HID * HID + 255) / 256, 256, 0, stream>>>(W, Wb, HID * HID);
    gemm_wmma_kernel<<<N_NODES / 16, 256, 0, stream>>>(hin, Wb, tmp_xw);
    hipMemsetAsync(hout, 0, (size_t)NH * sizeof(float), stream);
    aggregate_kernel<<<(TOT_E + 7) / 8, 256, 0, stream>>>(src, dst, dinv, tmp_xw, hout);
    bias_act_kernel<<<(NH + 255) / 256, 256, 0, stream>>>(hout, b, do_relu);
}

extern "C" void kernel_launch(void* const* d_in, const int* in_sizes, int n_in,
                              void* d_out, int out_size, void* d_ws, size_t ws_size,
                              hipStream_t stream) {
    const float* x     = (const float*)d_in[0];
    const int*   ei    = (const int*)d_in[1];     // [2, E] flat
    const int*   batch = (const int*)d_in[2];
    const float* W1 = (const float*)d_in[3];  const float* b1 = (const float*)d_in[4];
    const float* W2 = (const float*)d_in[5];  const float* b2 = (const float*)d_in[6];
    const float* W3 = (const float*)d_in[7];  const float* b3 = (const float*)d_in[8];
    const float* Wl = (const float*)d_in[9];  const float* bl = (const float*)d_in[10];
    float* out = (float*)d_out;

    const int* src = ei;
    const int* dst = ei + N_EDGES;

    float* tmp_xw = (float*)d_ws;            // NH
    float* hA     = tmp_xw + NH;             // NH
    float* hB     = hA + NH;                 // NH
    float* deg    = hB + NH;                 // N_NODES
    float* dinv   = deg + N_NODES;           // N_NODES
    float* pooled = dinv + N_NODES;          // N_GRAPHS*HID
    float* cnt    = pooled + N_GRAPHS * HID; // N_GRAPHS
    __bf16* Wb    = (__bf16*)(cnt + N_GRAPHS);

    // symmetric normalization (shared by all layers)
    hipMemsetAsync(deg, 0, N_NODES * sizeof(float), stream);
    deg_kernel<<<(TOT_E + 255) / 256, 256, 0, stream>>>(dst, deg);
    dinv_kernel<<<(N_NODES + 255) / 256, 256, 0, stream>>>(deg, dinv);

    run_layer(x,  W1, b1, tmp_xw, hA, Wb, src, dst, dinv, 1, stream);
    run_layer(hA, W2, b2, tmp_xw, hB, Wb, src, dst, dinv, 1, stream);
    run_layer(hB, W3, b3, tmp_xw, hA, Wb, src, dst, dinv, 0, stream);

    hipMemsetAsync(pooled, 0, (size_t)N_GRAPHS * HID * sizeof(float), stream);
    hipMemsetAsync(cnt, 0, N_GRAPHS * sizeof(float), stream);
    pool_kernel<<<(N_NODES + 7) / 8, 256, 0, stream>>>(hA, batch, pooled, cnt);
    head_kernel<<<(N_GRAPHS * N_CLASSES + 255) / 256, 256, 0, stream>>>(pooled, cnt, Wl, bl, out);
}
